// EGraphSAGE_86474871537827
// MI455X (gfx1250) — matrix operand order repaired and checked by reading, not compiled, not run on hardware
//
#include <hip/hip_runtime.h>
#include <cstdint>

#define N_NODES 50000
#define N_EDGES 600000
#define DFEAT   128

typedef _Float16 v16h __attribute__((ext_vector_type(16)));
typedef _Float16 v8h  __attribute__((ext_vector_type(8)));
typedef float    v8f  __attribute__((ext_vector_type(8)));

__device__ __forceinline__ float sigmoidf_(float x) {
    return 1.0f / (1.0f + __expf(-x));
}

// Build a v16h fragment from two 16-byte loads (kept in SSA form: no alloca).
__device__ __forceinline__ v16h ldfrag2(const _Float16* p0, const _Float16* p1) {
    v8h lo = *(const v8h*)p0;
    v8h hi = *(const v8h*)p1;
    return __builtin_shufflevector(lo, hi, 0, 1, 2, 3, 4, 5, 6, 7,
                                           8, 9, 10, 11, 12, 13, 14, 15);
}

// ---------------- zero-fill scratch accumulators ----------------
__global__ void k_zero(float* __restrict__ p, long n) {
    long i = (long)blockIdx.x * blockDim.x + threadIdx.x;
    long stride = (long)gridDim.x * blockDim.x;
    for (; i < n; i += stride) p[i] = 0.0f;
}

// ---------------- scatter edge_attr -> agg1[dst], counts[dst] ----------------
// 256 threads = 2 edges x 128 features
__global__ void k_scatter_edges(const float* __restrict__ eattr,
                                const long long* __restrict__ ei,
                                float* __restrict__ agg,
                                float* __restrict__ counts) {
    int slot = threadIdx.x >> 7;
    int t    = threadIdx.x & 127;
    long e   = (long)blockIdx.x * 2 + slot;
    if (e >= N_EDGES) return;
    int dst = (int)ei[N_EDGES + e];
    atomicAdd(&agg[(size_t)dst * DFEAT + t], eattr[(size_t)e * DFEAT + t]);
    if (t == 0) atomicAdd(&counts[dst], 1.0f);
}

// ---------------- scatter h1[src] -> agg2[dst] ----------------
__global__ void k_scatter_h1(const float* __restrict__ h1,
                             const long long* __restrict__ ei,
                             float* __restrict__ agg) {
    int slot = threadIdx.x >> 7;
    int t    = threadIdx.x & 127;
    long e   = (long)blockIdx.x * 2 + slot;
    if (e >= N_EDGES) return;
    int src = (int)ei[e];
    int dst = (int)ei[N_EDGES + e];
    atomicAdd(&agg[(size_t)dst * DFEAT + t], h1[(size_t)src * DFEAT + t]);
}

// ---------------- build X1 = f16([node_attr | agg1/cnt]) [N,256] ----------------
__global__ void k_build_x1(const float* __restrict__ node_attr,
                           const float* __restrict__ agg1,
                           const float* __restrict__ counts,
                           _Float16* __restrict__ X1) {
    int n = blockIdx.x, t = threadIdx.x;       // 256 threads
    float v;
    if (t < DFEAT) {
        v = node_attr[(size_t)n * DFEAT + t];
    } else {
        float c = fmaxf(counts[n], 1.0f);
        v = agg1[(size_t)n * DFEAT + (t - DFEAT)] / c;
    }
    X1[(size_t)n * 256 + t] = (_Float16)v;
}

// ---------------- build X2 = f16([h1 | agg2/cnt | mask*h1]) [N,384] ----------------
__global__ void k_build_x2(const float* __restrict__ h1,
                           const float* __restrict__ agg2,
                           const float* __restrict__ counts,
                           _Float16* __restrict__ X2) {
    int n = blockIdx.x, t = threadIdx.x;       // 384 threads
    float craw = counts[n];
    float c    = fmaxf(craw, 1.0f);
    float v;
    if (t < 128)       v = h1[(size_t)n * 128 + t];
    else if (t < 256)  v = agg2[(size_t)n * 128 + (t - 128)] / c;
    else               v = (craw > 0.0f) ? h1[(size_t)n * 128 + (t - 256)] : 0.0f;
    X2[(size_t)n * 384 + t] = (_Float16)v;
}

// ---------------- pack W [K,128] f32 -> WMMA B-fragment layout f16 ----------------
// Wp index d = (((kchunk*8 + coltile)*32 + lane)*16 + h)
// lane L holds column coltile*16 + (L&15), K = kchunk*32 + (h<8 ? 8*(L>>4) + h : 16 + 8*(L>>4) + (h-8))
__global__ void k_pack_w(const float* __restrict__ W, _Float16* __restrict__ Wp, int K) {
    int total = K * 128;
    int d = blockIdx.x * blockDim.x + threadIdx.x;
    if (d >= total) return;
    int h      = d & 15;
    int lane   = (d >> 4) & 31;
    int ct     = (d >> 9) & 7;
    int kchunk = d >> 12;
    int kk  = kchunk * 32 + ((h < 8) ? (8 * (lane >> 4) + h) : (16 + 8 * (lane >> 4) + (h - 8)));
    int col = ct * 16 + (lane & 15);
    Wp[d] = (_Float16)W[(size_t)kk * 128 + col];
}

// ---------------- H = sigmoid(X @ W + b), X [N,K] f16, W packed, H [N,128] f32 ----
// 128 threads = 4 waves; wave w owns col tiles 2w/2w+1: one shared A fragment
// feeds two WMMAs per K-chunk. 2-deep software pipeline with *named* fragments
// (pairs of K-chunks; chunk count is even for K=256 and K=384) so everything
// stays in VGPRs -- no runtime-indexed arrays, no alloca, no spills.
__global__ void __launch_bounds__(128)
k_gemm_sigmoid(const _Float16* __restrict__ X, const _Float16* __restrict__ Wp,
               const float* __restrict__ bias, float* __restrict__ H, int K) {
    const int lane = threadIdx.x & 31;
    const int w    = threadIdx.x >> 5;          // 0..3
    const int ct0  = 2 * w;                     // first col tile
    const int m0   = blockIdx.x * 16;
    const int hi   = lane >> 4;
    const int r16  = lane & 15;

    // A fragment (ISA 7.12.2 16-bit A layout): row = m0 + (lane&15),
    // halves 0..7 -> K = 32c + 8*hi.., halves 8..15 -> K = 32c + 16 + 8*hi..
    const _Float16* Xrow = X + (size_t)(m0 + r16) * K + 8 * hi;
    // B fragments: packed contiguous 16 halves per lane; chunk stride 4096 halves
    const _Float16* Bb0 = Wp + (((size_t)ct0 * 32 + lane) << 4);
    const _Float16* Bb1 = Wp + (((size_t)(ct0 + 1) * 32 + lane) << 4);

    const int nc = K >> 5;                      // number of 32-wide K chunks (even)
    v8f acc0 = {}, acc1 = {};

#define LD_A(c)  ldfrag2(Xrow + (c) * 32, Xrow + (c) * 32 + 16)
#define LD_B0(c) ldfrag2(Bb0 + ((size_t)(c) << 12), Bb0 + ((size_t)(c) << 12) + 8)
#define LD_B1(c) ldfrag2(Bb1 + ((size_t)(c) << 12), Bb1 + ((size_t)(c) << 12) + 8)
#define WMMA(A, B, C) __builtin_amdgcn_wmma_f32_16x16x32_f16( \
        false, (A), false, (B), (short)0, (C), false, false)

    v16h a0 = LD_A(0), p0 = LD_B0(0), q0 = LD_B1(0);
    for (int c = 0; c + 2 < nc; c += 2) {
        v16h a1 = LD_A(c + 1), p1 = LD_B0(c + 1), q1 = LD_B1(c + 1);
        __builtin_prefetch(Xrow + (c + 2) * 32, 0, 1);
        acc0 = WMMA(a0, p0, acc0);
        acc1 = WMMA(a0, q0, acc1);
        a0 = LD_A(c + 2); p0 = LD_B0(c + 2); q0 = LD_B1(c + 2);
        acc0 = WMMA(a1, p1, acc0);
        acc1 = WMMA(a1, q1, acc1);
    }
    {
        v16h a1 = LD_A(nc - 1), p1 = LD_B0(nc - 1), q1 = LD_B1(nc - 1);
        acc0 = WMMA(a0, p0, acc0);
        acc1 = WMMA(a0, q0, acc1);
        acc0 = WMMA(a1, p1, acc0);
        acc1 = WMMA(a1, q1, acc1);
    }
#undef LD_A
#undef LD_B0
#undef LD_B1
#undef WMMA

    // C/D layout: VGPR r of lane -> (M = r + 8*hi, N = lane&15)
    const int col0  = ct0 * 16 + r16;
    const float bc0 = bias[col0];
    const float bc1 = bias[col0 + 16];
    #pragma unroll
    for (int r = 0; r < 8; ++r) {
        int row = m0 + 8 * hi + r;
        H[(size_t)row * 128 + col0]      = sigmoidf_(acc0[r] + bc0);
        H[(size_t)row * 128 + col0 + 16] = sigmoidf_(acc1[r] + bc1);
    }
}

// ---------------- per-node sa = h2.Wf[:128], sb = h2.Wf[128:] ----------------
__global__ void k_node_dot(const float* __restrict__ h2, const float* __restrict__ Wf,
                           float* __restrict__ sa, float* __restrict__ sb) {
    int wv   = threadIdx.x >> 5;
    int lane = threadIdx.x & 31;
    int n = blockIdx.x * 8 + wv;
    if (n >= N_NODES) return;
    const float* row = h2 + (size_t)n * 128;
    float a = 0.f, b = 0.f;
    #pragma unroll
    for (int j = 0; j < 4; ++j) {
        int k = lane + 32 * j;
        float v = row[k];
        a += v * Wf[k];
        b += v * Wf[128 + k];
    }
    #pragma unroll
    for (int off = 16; off > 0; off >>= 1) {
        a += __shfl_xor(a, off, 32);
        b += __shfl_xor(b, off, 32);
    }
    if (lane == 0) { sa[n] = a; sb[n] = b; }
}

// ---------------- e2 = [h2[src] | h2[dst]], probs = sigmoid(sa+sb+bf) -----------
// 256 threads = 4 edges x 64 lanes; float4 copies (32 per half-row)
__global__ void k_edge_out(const float* __restrict__ h2,
                           const long long* __restrict__ ei,
                           const float* __restrict__ sa, const float* __restrict__ sb,
                           const float* __restrict__ bf,
                           float* __restrict__ out) {
    int slot = threadIdx.x >> 6;
    int t    = threadIdx.x & 63;
    long e   = (long)blockIdx.x * 4 + slot;
    if (e >= N_EDGES) return;
    int src = (int)ei[e];
    int dst = (int)ei[N_EDGES + e];
    float4* e2row = (float4*)(out + N_EDGES + (size_t)e * 256);
    if (t < 32) e2row[t] = ((const float4*)(h2 + (size_t)src * 128))[t];
    else        e2row[t] = ((const float4*)(h2 + (size_t)dst * 128))[t - 32];
    if (t == 0) out[e] = sigmoidf_(sa[src] + sb[dst] + bf[0]);
}

extern "C" void kernel_launch(void* const* d_in, const int* in_sizes, int n_in,
                              void* d_out, int out_size, void* d_ws, size_t ws_size,
                              hipStream_t stream) {
    const float*     edge_attr = (const float*)d_in[0];
    const float*     node_attr = (const float*)d_in[1];
    const float*     W1 = (const float*)d_in[2];
    const float*     b1 = (const float*)d_in[3];
    const float*     W2 = (const float*)d_in[4];
    const float*     b2 = (const float*)d_in[5];
    const float*     Wf = (const float*)d_in[6];
    const float*     bf = (const float*)d_in[7];
    const long long* ei = (const long long*)d_in[8];   // int64 [2,E]: src then dst
    float* out = (float*)d_out;

    // workspace layout (counts/agg1/agg2 contiguous so one zero-fill covers them)
    char* p = (char*)d_ws;
    float* counts = (float*)p; p += (size_t)N_NODES * 4;
    float* agg1   = (float*)p; p += (size_t)N_NODES * DFEAT * 4;
    float* agg2   = (float*)p; p += (size_t)N_NODES * DFEAT * 4;
    float* h1     = (float*)p; p += (size_t)N_NODES * DFEAT * 4;
    float* h2     = (float*)p; p += (size_t)N_NODES * DFEAT * 4;
    float* sa     = (float*)p; p += (size_t)N_NODES * 4;
    float* sb     = (float*)p; p += (size_t)N_NODES * 4;
    _Float16* X1  = (_Float16*)p; p += (size_t)N_NODES * 256 * 2;
    _Float16* X2  = (_Float16*)p; p += (size_t)N_NODES * 384 * 2;
    _Float16* W1p = (_Float16*)p; p += (size_t)256 * 128 * 2;
    _Float16* W2p = (_Float16*)p; p += (size_t)384 * 128 * 2;

    long zero_n = (long)N_NODES * (1 + 2 * DFEAT);
    k_zero<<<4096, 256, 0, stream>>>(counts, zero_n);
    k_pack_w<<<(256 * 128 + 255) / 256, 256, 0, stream>>>(W1, W1p, 256);
    k_pack_w<<<(384 * 128 + 255) / 256, 256, 0, stream>>>(W2, W2p, 384);

    k_scatter_edges<<<(N_EDGES + 1) / 2, 256, 0, stream>>>(edge_attr, ei, agg1, counts);
    k_build_x1<<<N_NODES, 256, 0, stream>>>(node_attr, agg1, counts, X1);
    k_gemm_sigmoid<<<N_NODES / 16, 128, 0, stream>>>(X1, W1p, b1, h1, 256);

    k_scatter_h1<<<(N_EDGES + 1) / 2, 256, 0, stream>>>(h1, ei, agg2);
    k_build_x2<<<N_NODES, 384, 0, stream>>>(h1, agg2, counts, X2);
    k_gemm_sigmoid<<<N_NODES / 16, 128, 0, stream>>>(X2, W2p, b2, h2, 384);

    k_node_dot<<<(N_NODES + 7) / 8, 256, 0, stream>>>(h2, Wf, sa, sb);
    k_edge_out<<<(N_EDGES + 3) / 4, 256, 0, stream>>>(h2, ei, sa, sb, bf, out);

    (void)in_sizes; (void)n_in; (void)out_size; (void)ws_size;
}
